// MultiGCN_relation_44959717655003
// MI455X (gfx1250) — matrix operand order repaired and checked by reading, not compiled, not run on hardware
//
#include <hip/hip_runtime.h>
#include <hip/hip_bf16.h>

#define NROW 2048
#define DIM  1600
#define HIDN 1000
#define HIDP 1024
#define TOPK 410
#define EPSF 2.2204460492503131e-16f
#define BN_SCALE 0.99999500003749977f  /* 1/sqrt(1+1e-5) */

typedef __attribute__((ext_vector_type(8)))  __bf16 bf16x8;
typedef __attribute__((ext_vector_type(16))) __bf16 bf16x16;
typedef __attribute__((ext_vector_type(8)))  float  f32x8;

#define WMMA_BF16(a, b, c) \
  __builtin_amdgcn_wmma_f32_16x16x32_bf16(false, (a), false, (b), (short)0, (c), false, false)

__device__ __forceinline__ unsigned short f2bf(float f) {
  unsigned u = __float_as_uint(f);
  unsigned r = u + 0x7FFFu + ((u >> 16) & 1u);   // round-to-nearest-even
  return (unsigned short)(r >> 16);
}

// A fragment (16x32 bf16): lane half 0 holds K {0..7,16..23}, half 1 holds {8..15,24..31}.
// p points at A_row + kk + half*8 ; elements 0..7 = p[0..7], 8..15 = p[16..23].
__device__ __forceinline__ bf16x16 load_a_frag(const unsigned short* p) {
  uint4 lo = *reinterpret_cast<const uint4*>(p);
  uint4 hi = *reinterpret_cast<const uint4*>(p + 16);
  bf16x8 a = __builtin_bit_cast(bf16x8, lo);
  bf16x8 b = __builtin_bit_cast(bf16x8, hi);
  bf16x16 r;
#pragma unroll
  for (int i = 0; i < 8; ++i) { r[i] = a[i]; r[i + 8] = b[i]; }
  return r;
}

// B fragment (32x16 bf16), column contiguous in K (NT form: Bt row = output column).
// lane half 0 holds K {0..7,16..23}, half 1 holds {8..15,24..31}; p = Bt_row + kk + half*8.
__device__ __forceinline__ bf16x16 load_b_frag(const unsigned short* p) {
  uint4 lo = *reinterpret_cast<const uint4*>(p);
  uint4 hi = *reinterpret_cast<const uint4*>(p + 16);
  bf16x8 a = __builtin_bit_cast(bf16x8, lo);
  bf16x8 b = __builtin_bit_cast(bf16x8, hi);
  bf16x16 r;
#pragma unroll
  for (int i = 0; i < 8; ++i) { r[i] = a[i]; r[i + 8] = b[i]; }
  return r;
}

// ---------------------------------------------------------------------------
// 1. Relation network: one block (64 threads) per sample -> sigma[n]
// ---------------------------------------------------------------------------
__global__ void relation_kernel(const float* __restrict__ feat,
                                const float* __restrict__ c1w, const float* __restrict__ c1b,
                                const float* __restrict__ g1,  const float* __restrict__ b1,
                                const float* __restrict__ c2w, const float* __restrict__ c2b,
                                const float* __restrict__ g2,  const float* __restrict__ b2,
                                const float* __restrict__ fc3w, const float* __restrict__ fc3b,
                                const float* __restrict__ fc4w, const float* __restrict__ fc4b,
                                float* __restrict__ sigma) {
  __shared__ float lx[64 * 25];   // input  [64,5,5]
  __shared__ float h1[64 * 25];   // conv1  [64,5,5]
  __shared__ float p1[64 * 9];    // pool1  [64,3,3]
  __shared__ float lc2[9];        // conv2  [1,3,3]
  __shared__ float v4[4];         // pool2  [1,2,2]
  __shared__ float h3[8];         // fc3
  const int n = blockIdx.x, tid = threadIdx.x;

  for (int i = tid; i < 64 * 25; i += 64) lx[i] = feat[(size_t)n * DIM + i];
  __syncthreads();

  // conv1 3x3 pad1 + BN + ReLU, output channel = tid
  {
    const int oc = tid;
    float acc[25];
#pragma unroll
    for (int p = 0; p < 25; ++p) acc[p] = c1b[oc];
    for (int ic = 0; ic < 64; ++ic) {
      float w[9];
#pragma unroll
      for (int t = 0; t < 9; ++t) w[t] = c1w[(size_t)(oc * 64 + ic) * 9 + t];
      const float* xin = &lx[ic * 25];
#pragma unroll
      for (int oh = 0; oh < 5; ++oh)
#pragma unroll
        for (int ow = 0; ow < 5; ++ow) {
          float s = acc[oh * 5 + ow];
#pragma unroll
          for (int kh = 0; kh < 3; ++kh)
#pragma unroll
            for (int kw = 0; kw < 3; ++kw) {
              int ih = oh + kh - 1, iw = ow + kw - 1;
              if (ih >= 0 && ih < 5 && iw >= 0 && iw < 5)
                s = fmaf(xin[ih * 5 + iw], w[kh * 3 + kw], s);
            }
          acc[oh * 5 + ow] = s;
        }
    }
    const float sc = g1[oc] * BN_SCALE, bb = b1[oc];
#pragma unroll
    for (int p = 0; p < 25; ++p) h1[oc * 25 + p] = fmaxf(acc[p] * sc + bb, 0.0f);
  }
  __syncthreads();

  // maxpool 2x2 stride2 pad1 : 5x5 -> 3x3
  {
    const int oc = tid;
#pragma unroll
    for (int ph = 0; ph < 3; ++ph)
#pragma unroll
      for (int pw = 0; pw < 3; ++pw) {
        float m = -3.0e38f;
#pragma unroll
        for (int dh = 0; dh < 2; ++dh)
#pragma unroll
          for (int dw = 0; dw < 2; ++dw) {
            int ih = 2 * ph - 1 + dh, iw = 2 * pw - 1 + dw;
            if (ih >= 0 && ih < 5 && iw >= 0 && iw < 5)
              m = fmaxf(m, h1[oc * 25 + ih * 5 + iw]);
          }
        p1[oc * 9 + ph * 3 + pw] = m;
      }
  }
  __syncthreads();

  // conv2 64->1 3x3 pad1 on 3x3 + BN + ReLU
  if (tid < 9) {
    const int oh = tid / 3, ow = tid % 3;
    float acc = c2b[0];
    for (int ic = 0; ic < 64; ++ic) {
#pragma unroll
      for (int kh = 0; kh < 3; ++kh)
#pragma unroll
        for (int kw = 0; kw < 3; ++kw) {
          int ih = oh + kh - 1, iw = ow + kw - 1;
          if (ih >= 0 && ih < 3 && iw >= 0 && iw < 3)
            acc = fmaf(p1[ic * 9 + ih * 3 + iw], c2w[(size_t)ic * 9 + kh * 3 + kw], acc);
        }
    }
    lc2[tid] = fmaxf(acc * (g2[0] * BN_SCALE) + b2[0], 0.0f);
  }
  __syncthreads();

  // maxpool pad1: 3x3 -> 2x2
  if (tid < 4) {
    const int ph = tid >> 1, pw = tid & 1;
    float m = -3.0e38f;
#pragma unroll
    for (int dh = 0; dh < 2; ++dh)
#pragma unroll
      for (int dw = 0; dw < 2; ++dw) {
        int ih = 2 * ph - 1 + dh, iw = 2 * pw - 1 + dw;
        if (ih >= 0 && ih < 3 && iw >= 0 && iw < 3) m = fmaxf(m, lc2[ih * 3 + iw]);
      }
    v4[ph * 2 + pw] = m;
  }
  __syncthreads();

  if (tid < 8) {
    float s = fc3b[tid];
#pragma unroll
    for (int i = 0; i < 4; ++i) s = fmaf(v4[i], fc3w[tid * 4 + i], s);
    h3[tid] = fmaxf(s, 0.0f);
  }
  __syncthreads();

  if (tid == 0) {
    float s = fc4b[0];
#pragma unroll
    for (int i = 0; i < 8; ++i) s = fmaf(h3[i], fc4w[i], s);
    sigma[n] = s;
  }
}

// ---------------------------------------------------------------------------
// 2. Row scale: f = feat/(sigma+eps) -> bf16, X -> bf16, sq[n] = sum f^2
// ---------------------------------------------------------------------------
__global__ void scale_kernel(const float* __restrict__ feat, const float* __restrict__ sigma,
                             unsigned short* __restrict__ fbf, unsigned short* __restrict__ xbf,
                             float* __restrict__ sq) {
  __shared__ float red[256];
  const int n = blockIdx.x, tid = threadIdx.x;
  const float inv = 1.0f / (sigma[n] + EPSF);
  float acc = 0.0f;
  for (int d = tid; d < DIM; d += 256) {
    float x = feat[(size_t)n * DIM + d];
    float f = x * inv;
    fbf[(size_t)n * DIM + d] = f2bf(f);
    xbf[(size_t)n * DIM + d] = f2bf(x);
    acc = fmaf(f, f, acc);
  }
  red[tid] = acc;
  __syncthreads();
  for (int s = 128; s > 0; s >>= 1) {
    if (tid < s) red[tid] += red[tid + s];
    __syncthreads();
  }
  if (tid == 0) sq[n] = red[0];
}

// 3. W -> bf16 TRANSPOSED: Wt[j][k] = W[k][j], j padded 1000 -> 1024
__global__ void wconv_kernel(const float* __restrict__ w, unsigned short* __restrict__ wbt) {
  int idx = blockIdx.x * 256 + threadIdx.x;                 // over HIDP*DIM
  int j = idx / DIM, k = idx % DIM;
  wbt[idx] = (j < HIDN) ? f2bf(w[(size_t)k * HIDN + j]) : (unsigned short)0;
}

// ---------------------------------------------------------------------------
// 4. Gram + exp epilogue: Ae = exp(-max(sq_i + sq_j - 2 f_i.f_j, 0))
//    NT form (B cols are rows of f). 32x32 per wave; block = 8 waves = 64x128.
// ---------------------------------------------------------------------------
__global__ void gram_kernel(const unsigned short* __restrict__ fbf,
                            const float* __restrict__ sq, float* __restrict__ Ae) {
  const int lane = threadIdx.x & 31, wid = threadIdx.x >> 5;
  const int wm = wid & 1, wn = wid >> 1;
  const int tile_m = blockIdx.x * 64 + wm * 32;
  const int tile_n = blockIdx.y * 128 + wn * 32;
  const int half = lane >> 4, l15 = lane & 15;
  const unsigned short* ap0 = fbf + (size_t)(tile_m + l15) * DIM + half * 8;
  const unsigned short* ap1 = fbf + (size_t)(tile_m + 16 + l15) * DIM + half * 8;
  const unsigned short* bp0 = fbf + (size_t)(tile_n + l15) * DIM + half * 8;
  const unsigned short* bp1 = fbf + (size_t)(tile_n + 16 + l15) * DIM + half * 8;
  f32x8 c00 = {0, 0, 0, 0, 0, 0, 0, 0}, c01 = c00, c10 = c00, c11 = c00;
  for (int kk = 0; kk < DIM; kk += 32) {
    __builtin_prefetch(ap0 + kk + 256, 0, 0);
    bf16x16 a0 = load_a_frag(ap0 + kk);
    bf16x16 a1 = load_a_frag(ap1 + kk);
    bf16x16 b0 = load_b_frag(bp0 + kk);
    bf16x16 b1 = load_b_frag(bp1 + kk);
    c00 = WMMA_BF16(a0, b0, c00);
    c01 = WMMA_BF16(a0, b1, c01);
    c10 = WMMA_BF16(a1, b0, c10);
    c11 = WMMA_BF16(a1, b1, c11);
  }
  const int oc0 = tile_n + l15, oc1 = oc0 + 16;
  const float s0 = sq[oc0], s1 = sq[oc1];
#pragma unroll
  for (int r = 0; r < 8; ++r) {
    int orow0 = tile_m + half * 8 + r, orow1 = orow0 + 16;
    float sm0 = sq[orow0], sm1 = sq[orow1];
    Ae[(size_t)orow0 * NROW + oc0] = __expf(-fmaxf(sm0 + s0 - 2.0f * c00[r], 0.0f));
    Ae[(size_t)orow0 * NROW + oc1] = __expf(-fmaxf(sm0 + s1 - 2.0f * c01[r], 0.0f));
    Ae[(size_t)orow1 * NROW + oc0] = __expf(-fmaxf(sm1 + s0 - 2.0f * c10[r], 0.0f));
    Ae[(size_t)orow1 * NROW + oc1] = __expf(-fmaxf(sm1 + s1 - 2.0f * c11[r], 0.0f));
  }
}

// ---------------------------------------------------------------------------
// 5. Per-row exact K-th largest (bit bisection, Ae > 0) + degree
// ---------------------------------------------------------------------------
__global__ void topk_kernel(const float* __restrict__ Ae,
                            unsigned* __restrict__ thr, float* __restrict__ invd) {
  __shared__ unsigned su[NROW];
  __shared__ unsigned scnt;
  __shared__ float ssum;
  const int row = blockIdx.x, tid = threadIdx.x;
  for (int j = tid; j < NROW; j += 256) su[j] = __float_as_uint(Ae[(size_t)row * NROW + j]);
  __syncthreads();
  unsigned T = 0;
  for (int bit = 30; bit >= 0; --bit) {
    unsigned cand = T | (1u << bit);
    if (tid == 0) scnt = 0;
    __syncthreads();
    unsigned c = 0;
    for (int j = tid; j < NROW; j += 256) c += (su[j] >= cand) ? 1u : 0u;
    atomicAdd(&scnt, c);
    __syncthreads();
    if (scnt >= TOPK) T = cand;   // uniform across block
    __syncthreads();
  }
  if (tid == 0) ssum = 0.0f;
  __syncthreads();
  float ls = 0.0f;
  for (int j = tid; j < NROW; j += 256)
    if (su[j] >= T && j != row) ls += __uint_as_float(su[j]);
  atomicAdd(&ssum, ls);
  __syncthreads();
  if (tid == 0) { thr[row] = T; invd[row] = rsqrtf(ssum + 2.0f); }  // (1+S)+1
}

// 6. An (normalized adjacency) -> bf16
__global__ void an_kernel(const float* __restrict__ Ae, const unsigned* __restrict__ thr,
                          const float* __restrict__ invd, unsigned short* __restrict__ anbf) {
  const int i = blockIdx.x;
  const int j = blockIdx.y * 256 + threadIdx.x;
  float v;
  if (j == i) {
    v = invd[i] * invd[i];
  } else {
    float ae = Ae[(size_t)i * NROW + j];
    v = (__float_as_uint(ae) >= thr[i]) ? ae * invd[i] * invd[j] : 0.0f;
  }
  anbf[(size_t)i * NROW + j] = f2bf(v);
}

// ---------------------------------------------------------------------------
// 7. bf16 WMMA GEMM, NT form: C[M,N] = A[M,K] x Bt[N,K]^T, fp32 acc.
//    lda == ldbt == Kd for all uses. 32x32 per wave; block = 8 waves = 64x128.
//    grid = (M/64, N/128). Optionally writes a TRANSPOSED bf16 copy
//    Cbt[N][ldcbt] (Cbt[col][row]) so it can feed the next NT GEMM as Bt.
// ---------------------------------------------------------------------------
__global__ void gemm_nt_kernel(const unsigned short* __restrict__ A,
                               const unsigned short* __restrict__ Bt,
                               float* __restrict__ C, unsigned short* __restrict__ Cbt,
                               int ldc, int ldcbt, int Kd) {
  const int lane = threadIdx.x & 31, wid = threadIdx.x >> 5;
  const int wm = wid & 1, wn = wid >> 1;
  const int tile_m = blockIdx.x * 64 + wm * 32;
  const int tile_n = blockIdx.y * 128 + wn * 32;
  const int half = lane >> 4, l15 = lane & 15;
  const unsigned short* ap0 = A + (size_t)(tile_m + l15) * Kd + half * 8;
  const unsigned short* ap1 = A + (size_t)(tile_m + 16 + l15) * Kd + half * 8;
  const unsigned short* bp0 = Bt + (size_t)(tile_n + l15) * Kd + half * 8;
  const unsigned short* bp1 = Bt + (size_t)(tile_n + 16 + l15) * Kd + half * 8;
  f32x8 c00 = {0, 0, 0, 0, 0, 0, 0, 0}, c01 = c00, c10 = c00, c11 = c00;
  for (int kk = 0; kk < Kd; kk += 32) {
    __builtin_prefetch(ap0 + kk + 256, 0, 0);
    bf16x16 a0 = load_a_frag(ap0 + kk);
    bf16x16 a1 = load_a_frag(ap1 + kk);
    bf16x16 b0 = load_b_frag(bp0 + kk);
    bf16x16 b1 = load_b_frag(bp1 + kk);
    c00 = WMMA_BF16(a0, b0, c00);
    c01 = WMMA_BF16(a0, b1, c01);
    c10 = WMMA_BF16(a1, b0, c10);
    c11 = WMMA_BF16(a1, b1, c11);
  }
  const int oc0 = tile_n + l15, oc1 = oc0 + 16;
#pragma unroll
  for (int r = 0; r < 8; ++r) {
    int orow0 = tile_m + half * 8 + r, orow1 = orow0 + 16;
    C[(size_t)orow0 * ldc + oc0] = c00[r];
    C[(size_t)orow0 * ldc + oc1] = c01[r];
    C[(size_t)orow1 * ldc + oc0] = c10[r];
    C[(size_t)orow1 * ldc + oc1] = c11[r];
    if (Cbt) {
      Cbt[(size_t)oc0 * ldcbt + orow0] = f2bf(c00[r]);
      Cbt[(size_t)oc1 * ldcbt + orow0] = f2bf(c01[r]);
      Cbt[(size_t)oc0 * ldcbt + orow1] = f2bf(c10[r]);
      Cbt[(size_t)oc1 * ldcbt + orow1] = f2bf(c11[r]);
    }
  }
}

// ---------------------------------------------------------------------------
// 8. Epilogue: out = relu((a0*XW + a1*Y1 + a2*Y2 + gcn_b) * bn_g/sqrt(1+eps) + bn_b)
// ---------------------------------------------------------------------------
__global__ void final_kernel(const float* __restrict__ XW, const float* __restrict__ Y1,
                             const float* __restrict__ Y2,
                             const float* __restrict__ gcn_b,
                             const float* __restrict__ bn_g, const float* __restrict__ bn_b,
                             const float* __restrict__ aifa2, const float* __restrict__ aifa3,
                             float* __restrict__ out) {
  const int i = blockIdx.x;
  const int j = blockIdx.y * 256 + threadIdx.x;
  if (j >= HIDN) return;
  const float e2 = __expf(aifa2[0]), e3 = __expf(aifa3[0]);
  const float inv = 1.0f / (1.0f + e2 + e3);
  const float w0 = inv, w1 = e2 * inv, w2 = e3 * inv;
  const size_t p = (size_t)i * HIDP + j;
  float x = w0 * XW[p] + w1 * Y1[p] + w2 * Y2[p] + gcn_b[j];
  x = x * (bn_g[j] * BN_SCALE) + bn_b[j];
  out[(size_t)i * HIDN + j] = fmaxf(x, 0.0f);
}

// ---------------------------------------------------------------------------
extern "C" void kernel_launch(void* const* d_in, const int* in_sizes, int n_in,
                              void* d_out, int out_size, void* d_ws, size_t ws_size,
                              hipStream_t stream) {
  const float* feat  = (const float*)d_in[0];
  const float* c1w   = (const float*)d_in[1];
  const float* c1b   = (const float*)d_in[2];
  const float* g1    = (const float*)d_in[3];
  const float* b1    = (const float*)d_in[4];
  const float* c2w   = (const float*)d_in[5];
  const float* c2b   = (const float*)d_in[6];
  const float* g2    = (const float*)d_in[7];
  const float* b2    = (const float*)d_in[8];
  const float* fc3w  = (const float*)d_in[9];
  const float* fc3b  = (const float*)d_in[10];
  const float* fc4w  = (const float*)d_in[11];
  const float* fc4b  = (const float*)d_in[12];
  const float* aifa2 = (const float*)d_in[13];
  const float* aifa3 = (const float*)d_in[14];
  const float* gcn_w = (const float*)d_in[15];
  const float* gcn_b = (const float*)d_in[16];
  const float* bn2_g = (const float*)d_in[17];
  const float* bn2_b = (const float*)d_in[18];
  float* out = (float*)d_out;
  char* ws = (char*)d_ws;

  // workspace layout (bytes), with aliasing of dead regions; peak ~54.2 MB
  const size_t SZ_FBF = (size_t)NROW * DIM * 2;       // 6,553,600
  const size_t off_fbf = 0;
  const size_t off_xbf = SZ_FBF;
  const size_t off_wbt = 2 * SZ_FBF;
  const size_t off_ae  = off_wbt + (size_t)HIDP * DIM * 2;
  const size_t off_anb = off_ae + (size_t)NROW * NROW * 4;
  const size_t off_xw  = off_anb + (size_t)NROW * NROW * 2;
  const size_t off_xwt = off_xw + (size_t)NROW * HIDP * 4;
  const size_t off_y1  = off_ae;                                   // Ae dead after an_kernel
  const size_t off_y1t = off_ae + (size_t)NROW * HIDP * 4;
  const size_t off_y2  = 0;                                        // fbf/xbf dead by then
  const size_t off_sm  = off_xwt + (size_t)NROW * HIDP * 2;

  unsigned short* fbf  = (unsigned short*)(ws + off_fbf);
  unsigned short* xbf  = (unsigned short*)(ws + off_xbf);
  unsigned short* wbt  = (unsigned short*)(ws + off_wbt);   // [HIDP][DIM]
  float*          Ae   = (float*)(ws + off_ae);
  unsigned short* anb  = (unsigned short*)(ws + off_anb);   // [NROW][NROW]
  float*          XW   = (float*)(ws + off_xw);             // [NROW][HIDP]
  unsigned short* XWt  = (unsigned short*)(ws + off_xwt);   // [HIDP][NROW]
  float*          Y1   = (float*)(ws + off_y1);
  unsigned short* Y1t  = (unsigned short*)(ws + off_y1t);   // [HIDP][NROW]
  float*          Y2   = (float*)(ws + off_y2);
  float*    sigma = (float*)(ws + off_sm);
  float*    sq    = (float*)(ws + off_sm + NROW * 4);
  unsigned* thr   = (unsigned*)(ws + off_sm + 2 * NROW * 4);
  float*    invd  = (float*)(ws + off_sm + 3 * NROW * 4);

  relation_kernel<<<NROW, 64, 0, stream>>>(feat, c1w, c1b, g1, b1, c2w, c2b, g2, b2,
                                           fc3w, fc3b, fc4w, fc4b, sigma);
  scale_kernel<<<NROW, 256, 0, stream>>>(feat, sigma, fbf, xbf, sq);
  wconv_kernel<<<(HIDP * DIM) / 256, 256, 0, stream>>>(gcn_w, wbt);
  gram_kernel<<<dim3(NROW / 64, NROW / 128), 256, 0, stream>>>(fbf, sq, Ae);
  topk_kernel<<<NROW, 256, 0, stream>>>(Ae, thr, invd);
  an_kernel<<<dim3(NROW, NROW / 256), 256, 0, stream>>>(Ae, thr, invd, anb);
  // XW = X @ W            (A = xbf [NROW][DIM],  Bt = wbt [HIDP][DIM],  K = DIM)
  gemm_nt_kernel<<<dim3(NROW / 64, HIDP / 128), 256, 0, stream>>>(xbf, wbt, XW, XWt,
                                                                  HIDP, NROW, DIM);
  // Y1 = An @ XW          (A = anb [NROW][NROW], Bt = XWt [HIDP][NROW], K = NROW)
  gemm_nt_kernel<<<dim3(NROW / 64, HIDP / 128), 256, 0, stream>>>(anb, XWt, Y1, Y1t,
                                                                  HIDP, NROW, NROW);
  // Y2 = An @ Y1
  gemm_nt_kernel<<<dim3(NROW / 64, HIDP / 128), 256, 0, stream>>>(anb, Y1t, Y2,
                                                                  (unsigned short*)nullptr,
                                                                  HIDP, NROW, NROW);
  final_kernel<<<dim3(NROW, (HIDN + 255) / 256), 256, 0, stream>>>(XW, Y1, Y2, gcn_b,
                                                                   bn2_g, bn2_b, aifa2, aifa3,
                                                                   out);
  (void)in_sizes; (void)n_in; (void)out_size; (void)ws_size;
}
